// SAKELayer_76081050681986
// MI455X (gfx1250) — compile-verified, matrix operand order checked by reading, not run on hardware
//
#include <hip/hip_runtime.h>
#include <hip/hip_bf16.h>

#define BATCH 16
#define NPT   256
#define FD    64
#define NCOEF 32
#define IPB   4
#define THREADS 256
#define NWAVE 8
#define EPSF 1e-10f

typedef _Float16 v16h __attribute__((ext_vector_type(16)));
typedef _Float16 v8h  __attribute__((ext_vector_type(8)));
typedef float    v8f  __attribute__((ext_vector_type(8)));

union Frag16 { v16h v; v8h h[2]; };

__device__ __forceinline__ float fast_rcp(float x) { return __builtin_amdgcn_rcpf(x); }
__device__ __forceinline__ float silu_f(float v) {
    return v * fast_rcp(1.0f + __expf(-v));          // v_exp + v_rcp, no IEEE div chain
}
__device__ __forceinline__ float sigm_f(float v) {
    return fast_rcp(1.0f + __expf(-v));
}

// CDNA5 wave32 16-bit matrix fragment load (ISA 7.12.2):
// lane<16 : row = lane,    K = {kb..kb+7, kb+16..kb+23} with kb = 0
// lane>=16: row = lane-16, K starts at kb = 8
// rows are contiguous 64-half (128B) runs in LDS -> two ds_load_b128 per fragment.
__device__ __forceinline__ v16h load_frag(const _Float16* base, int row, int kc) {
    Frag16 f;
    const int kb = kc * 32 + ((threadIdx.x & 16) ? 8 : 0);
    f.h[0] = *(const v8h*)(base + row * FD + kb);
    f.h[1] = *(const v8h*)(base + row * FD + kb + 16);
    return f.v;
}

__device__ __forceinline__ v8f wmma_f16(v16h a, v16h b, v8f c) {
    return __builtin_amdgcn_wmma_f32_16x16x32_f16(false, a, false, b, (short)0, c,
                                                  false, false);
}

// f16-row (v8h chunks) dot f32-vector from LDS: 8x ds_load_b128 instead of 64x u16
__device__ __forceinline__ float dot64_h(const _Float16* row, const float* wvec) {
    float s = 0.0f;
    #pragma unroll
    for (int fb = 0; fb < FD; fb += 8) {
        const v8h hh = *(const v8h*)(row + fb);
        #pragma unroll
        for (int e = 0; e < 8; ++e) s += (float)hh[e] * wvec[fb + e];
    }
    return s;
}

// ---------------------------------------------------------------------------
// Kernel 1: node-level precompute  P = h @ ew1[0:64], Q = h @ ew1[64:128]
// ---------------------------------------------------------------------------
__global__ __launch_bounds__(64) void sake_prep_pq(const float* __restrict__ h,
                                                   const float* __restrict__ ew1,
                                                   float* __restrict__ P,
                                                   float* __restrict__ Q) {
    const int node = blockIdx.x;
    const int f = threadIdx.x;
    __shared__ float hr[FD];
    hr[f] = h[node * FD + f];
    __syncthreads();
    float p = 0.0f, q = 0.0f;
    for (int k = 0; k < FD; ++k) {
        const float hv = hr[k];
        p += hv * ew1[k * FD + f];
        q += hv * ew1[(FD + k) * FD + f];
    }
    P[node * FD + f] = p;
    Q[node * FD + f] = q;
}

// ---------------------------------------------------------------------------
// Kernel 2: fused edge kernel. Block = 4 i-nodes x all 256 j. 8 wave32s.
// ---------------------------------------------------------------------------
__global__ __launch_bounds__(THREADS) void sake_edge_kernel(
    const float* __restrict__ x,
    const float* __restrict__ ew1, const float* __restrict__ eb1,
    const float* __restrict__ ew2, const float* __restrict__ eb2,
    const float* __restrict__ aw,  const float* __restrict__ ab,
    const float* __restrict__ cw1, const float* __restrict__ cb1,
    const float* __restrict__ cw2, const float* __restrict__ cb2,
    const float* __restrict__ xw1, const float* __restrict__ xb1,
    const float* __restrict__ xw2,
    const float* __restrict__ P,  const float* __restrict__ Q,
    float* __restrict__ hagg_g, float* __restrict__ comb_g,
    float* __restrict__ xout_g)
{
    __shared__ __align__(16) _Float16 ew2T[FD * FD];   // [n][k] transposed, f16
    __shared__ __align__(16) _Float16 cw1T[FD * FD];
    __shared__ __align__(16) _Float16 xw1T[FD * FD];
    __shared__ __align__(16) _Float16 cw2T[NCOEF * FD];
    __shared__ __align__(16) _Float16 buf[NWAVE][16 * FD];   // per-wave activation tile
    __shared__ float Pi[IPB][FD];
    __shared__ float xi[IPB][3];
    __shared__ float eb2s[FD], cb1s[FD], xb1s[FD], awf[FD], xw2f[FD], w128f[FD];
    __shared__ float cb2s[NCOEF];
    __shared__ float hagg[IPB][FD];
    __shared__ float comb[IPB][NCOEF * 3];
    __shared__ float dxs[IPB][3];
    __shared__ float gN[NWAVE][16];
    __shared__ float gD[NWAVE][16][3];
    __shared__ float gU[NWAVE][16][3];
    __shared__ float attv[NWAVE][16];
    __shared__ float ab0;

    const int tid  = threadIdx.x;
    const int lane = tid & 31;
    const int wv   = tid >> 5;
    const int nb0  = blockIdx.x * IPB;        // first global node (b*N + i0)
    const int bN   = (blockIdx.x >> 6) << 8;  // b * N  (64 blocks per batch)
    const int m    = lane & 15;

    // ---- stage weights (transposed f16) + biases ----
    for (int u = tid; u < FD * FD; u += THREADS) {
        const int k = u & 63, n = u >> 6;
        ew2T[n * FD + k] = (_Float16)ew2[k * FD + n];
        cw1T[n * FD + k] = (_Float16)cw1[k * FD + n];
        xw1T[n * FD + k] = (_Float16)xw1[k * FD + n];
    }
    for (int u = tid; u < NCOEF * FD; u += THREADS) {
        const int k = u & 63, n = u >> 6;
        cw2T[n * FD + k] = (_Float16)cw2[k * NCOEF + n];
    }
    if (tid < FD) {
        eb2s[tid]  = eb2[tid];
        cb1s[tid]  = cb1[tid];
        xb1s[tid]  = xb1[tid];
        awf[tid]   = aw[tid];
        xw2f[tid]  = xw2[tid];
        w128f[tid] = ew1[128 * FD + tid];   // ew1 row for the norm feature
    }
    if (tid < NCOEF) cb2s[tid] = cb2[tid];
    if (tid == 0) ab0 = ab[0];
    for (int u = tid; u < IPB * FD; u += THREADS) {
        const int il = u >> 6, f = u & 63;
        Pi[il][f]   = P[(nb0 + il) * FD + f] + eb1[f];
        hagg[il][f] = 0.0f;
    }
    for (int u = tid; u < IPB * NCOEF * 3; u += THREADS)
        comb[u / (NCOEF * 3)][u % (NCOEF * 3)] = 0.0f;
    if (tid < IPB * 3) {
        dxs[tid / 3][tid % 3] = 0.0f;
        xi[tid / 3][tid % 3]  = x[(nb0 + tid / 3) * 3 + tid % 3];
    }
    __syncthreads();

    // ---- tile loop: 4 i-nodes x 16 j-tiles = 64 tiles, 8 per wave (uniform) ----
    for (int t = wv; t < IPB * 16; t += NWAVE) {
        const int il = t >> 4;
        const int jb = (t & 15) << 4;

        // geometry for the 16 j's of this tile (lanes 0..15)
        if (lane < 16) {
            const int j = jb + m;
            const float d0 = xi[il][0] - x[(bN + j) * 3 + 0];
            const float d1 = xi[il][1] - x[(bN + j) * 3 + 1];
            const float d2 = xi[il][2] - x[(bN + j) * 3 + 2];
            const float nn  = __builtin_amdgcn_sqrtf(d0 * d0 + d1 * d1 + d2 * d2 + EPSF);
            const float inv = fast_rcp(nn + EPSF);
            gN[wv][m] = nn;
            gD[wv][m][0] = d0;        gD[wv][m][1] = d1;        gD[wv][m][2] = d2;
            gU[wv][m][0] = d0 * inv;  gU[wv][m][1] = d1 * inv;  gU[wv][m][2] = d2 * inv;
        }
        __syncthreads();

        // edge layer 1: act1 = silu(P_i + Q_j + norm*w128 + eb1) -> f16 tile
        for (int u = lane; u < 16 * FD; u += 32) {
            const int mm = u >> 6, f = u & 63;
            const float a = Pi[il][f] + Q[(bN + jb + mm) * FD + f] + gN[wv][mm] * w128f[f];
            buf[wv][mm * FD + f] = (_Float16)silu_f(a);
        }
        __syncthreads();

        // edge layer 2: h_e2 = silu(act1 @ ew2 + eb2)
        {
            const v16h a0 = load_frag(&buf[wv][0], m, 0);
            const v16h a1 = load_frag(&buf[wv][0], m, 1);
            const int mrow = ((lane >> 4) & 1) * 8;
            #pragma unroll
            for (int nc = 0; nc < 4; ++nc) {
                const int n = m + 16 * nc;
                v8f acc = {};
                acc = wmma_f16(a0, load_frag(ew2T, n, 0), acc);
                acc = wmma_f16(a1, load_frag(ew2T, n, 1), acc);
                #pragma unroll
                for (int r = 0; r < 8; ++r)
                    buf[wv][(mrow + r) * FD + n] = (_Float16)silu_f(acc[r] + eb2s[n]);
            }
        }
        __syncthreads();

        // sigmoid attention gate per edge
        if (lane < 16) {
            const float s = ab0 + dot64_h(&buf[wv][m * FD], awf);
            attv[wv][m] = sigm_f(s);
        }
        __syncthreads();

        // h_agg += sum_j att*h_e2
        for (int f = lane; f < FD; f += 32) {
            float s = 0.0f;
            for (int mm = 0; mm < 16; ++mm)
                s += attv[wv][mm] * (float)buf[wv][mm * FD + f];
            atomicAdd(&hagg[il][f], s);
        }

        // gated A fragments (per-lane scalar: all of a lane's A rows share one M)
        const _Float16 attm = (_Float16)attv[wv][m];
        const v16h g0 = load_frag(&buf[wv][0], m, 0) * attm;
        const v16h g1 = load_frag(&buf[wv][0], m, 1) * attm;
        __syncthreads();

        // coordinate branch: transH = silu(h_e @ xw1 + xb1)
        {
            const int mrow = ((lane >> 4) & 1) * 8;
            #pragma unroll
            for (int nc = 0; nc < 4; ++nc) {
                const int n = m + 16 * nc;
                v8f acc = {};
                acc = wmma_f16(g0, load_frag(xw1T, n, 0), acc);
                acc = wmma_f16(g1, load_frag(xw1T, n, 1), acc);
                #pragma unroll
                for (int r = 0; r < 8; ++r)
                    buf[wv][(mrow + r) * FD + n] = (_Float16)silu_f(acc[r] + xb1s[n]);
            }
        }
        __syncthreads();
        if (lane < 16) {
            const float s = dot64_h(&buf[wv][m * FD], xw2f);
            atomicAdd(&dxs[il][0], gD[wv][m][0] * s);
            atomicAdd(&dxs[il][1], gD[wv][m][1] * s);
            atomicAdd(&dxs[il][2], gD[wv][m][2] * s);
        }
        __syncthreads();

        // coefficient branch: t2 = silu(h_e @ cw1 + cb1)
        {
            const int mrow = ((lane >> 4) & 1) * 8;
            #pragma unroll
            for (int nc = 0; nc < 4; ++nc) {
                const int n = m + 16 * nc;
                v8f acc = {};
                acc = wmma_f16(g0, load_frag(cw1T, n, 0), acc);
                acc = wmma_f16(g1, load_frag(cw1T, n, 1), acc);
                #pragma unroll
                for (int r = 0; r < 8; ++r)
                    buf[wv][(mrow + r) * FD + n] = (_Float16)silu_f(acc[r] + cb1s[n]);
            }
        }
        __syncthreads();

        // coeff = t2 @ cw2 + cb2 ; comb[c][d] += coeff * unit
        {
            const v16h c0 = load_frag(&buf[wv][0], m, 0);
            const v16h c1 = load_frag(&buf[wv][0], m, 1);
            const int mrow = ((lane >> 4) & 1) * 8;
            #pragma unroll
            for (int nc = 0; nc < 2; ++nc) {
                const int c = m + 16 * nc;
                v8f acc = {};
                acc = wmma_f16(c0, load_frag(cw2T, c, 0), acc);
                acc = wmma_f16(c1, load_frag(cw2T, c, 1), acc);
                #pragma unroll
                for (int r = 0; r < 8; ++r) {
                    const float cf = acc[r] + cb2s[c];
                    const int mm = mrow + r;
                    atomicAdd(&comb[il][c * 3 + 0], cf * gU[wv][mm][0]);
                    atomicAdd(&comb[il][c * 3 + 1], cf * gU[wv][mm][1]);
                    atomicAdd(&comb[il][c * 3 + 2], cf * gU[wv][mm][2]);
                }
            }
        }
        __syncthreads();
    }

    __syncthreads();
    for (int u = tid; u < IPB * FD; u += THREADS)
        hagg_g[nb0 * FD + u] = hagg[u >> 6][u & 63];
    for (int u = tid; u < IPB * NCOEF * 3; u += THREADS)
        comb_g[nb0 * NCOEF * 3 + u] = comb[u / (NCOEF * 3)][u % (NCOEF * 3)];
    if (tid < IPB * 3) {
        const int il = tid / 3, d = tid % 3;
        xout_g[(nb0 + il) * 3 + d] = xi[il][d] + dxs[il][d] * (1.0f / (float)NPT);
    }
}

// ---------------------------------------------------------------------------
// Kernel 3: node-level post_norm MLP + node MLP + residual (tiny, VALU)
// ---------------------------------------------------------------------------
__global__ __launch_bounds__(64) void sake_node_kernel(
    const float* __restrict__ h, const float* __restrict__ hagg,
    const float* __restrict__ comb,
    const float* __restrict__ pw1, const float* __restrict__ pb1,
    const float* __restrict__ pw2, const float* __restrict__ pb2,
    const float* __restrict__ nw1, const float* __restrict__ nb1,
    const float* __restrict__ nw2, const float* __restrict__ nb2,
    float* __restrict__ hout)
{
    const int node = blockIdx.x;
    const int f = threadIdx.x;
    __shared__ float cn[NCOEF], tb[FD], nin[2 * FD + FD], u[FD];

    if (f < NCOEF) {
        const float* c = comb + node * NCOEF * 3 + f * 3;
        cn[f] = c[0] * c[0] + c[1] * c[1] + c[2] * c[2];
    }
    nin[f]      = h[node * FD + f];
    nin[FD + f] = hagg[node * FD + f];
    __syncthreads();

    float s = pb1[f];
    for (int k = 0; k < NCOEF; ++k) s += cn[k] * pw1[k * FD + f];
    tb[f] = silu_f(s);
    __syncthreads();

    s = pb2[f];
    for (int k = 0; k < FD; ++k) s += tb[k] * pw2[k * FD + f];
    nin[2 * FD + f] = s;   // h_comb
    __syncthreads();

    s = nb1[f];
    for (int k = 0; k < 3 * FD; ++k) s += nin[k] * nw1[k * FD + f];
    u[f] = silu_f(s);
    __syncthreads();

    s = nb2[f];
    for (int k = 0; k < FD; ++k) s += u[k] * nw2[k * FD + f];
    hout[node * FD + f] = h[node * FD + f] + s;
}

// ---------------------------------------------------------------------------
extern "C" void kernel_launch(void* const* d_in, const int* in_sizes, int n_in,
                              void* d_out, int out_size, void* d_ws, size_t ws_size,
                              hipStream_t stream) {
    (void)in_sizes; (void)n_in; (void)out_size; (void)ws_size;
    const float* h   = (const float*)d_in[0];
    const float* x   = (const float*)d_in[1];
    const float* ew1 = (const float*)d_in[2];
    const float* eb1 = (const float*)d_in[3];
    const float* ew2 = (const float*)d_in[4];
    const float* eb2 = (const float*)d_in[5];
    const float* aw  = (const float*)d_in[6];
    const float* ab  = (const float*)d_in[7];
    const float* cw1 = (const float*)d_in[8];
    const float* cb1 = (const float*)d_in[9];
    const float* cw2 = (const float*)d_in[10];
    const float* cb2 = (const float*)d_in[11];
    const float* pw1 = (const float*)d_in[12];
    const float* pb1 = (const float*)d_in[13];
    const float* pw2 = (const float*)d_in[14];
    const float* pb2 = (const float*)d_in[15];
    const float* nw1 = (const float*)d_in[16];
    const float* nb1 = (const float*)d_in[17];
    const float* nw2 = (const float*)d_in[18];
    const float* nb2 = (const float*)d_in[19];
    const float* xw1 = (const float*)d_in[20];
    const float* xb1 = (const float*)d_in[21];
    const float* xw2 = (const float*)d_in[22];

    const int NNODES = BATCH * NPT;                  // 4096
    float* ws   = (float*)d_ws;
    float* P    = ws;                                // 262144 f32
    float* Qb   = ws + (size_t)NNODES * FD;          // 262144 f32
    float* hag  = Qb + (size_t)NNODES * FD;          // 262144 f32
    float* cmb  = hag + (size_t)NNODES * FD;         // 393216 f32
    float* hout = (float*)d_out;                     // [4096,64]
    float* xout = hout + (size_t)NNODES * FD;        // [4096,3]

    sake_prep_pq<<<NNODES, 64, 0, stream>>>(h, ew1, P, Qb);
    sake_edge_kernel<<<NNODES / IPB, THREADS, 0, stream>>>(
        x, ew1, eb1, ew2, eb2, aw, ab, cw1, cb1, cw2, cb2,
        xw1, xb1, xw2, P, Qb, hag, cmb, xout);
    sake_node_kernel<<<NNODES, 64, 0, stream>>>(
        h, hag, cmb, pw1, pb1, pw2, pb2, nw1, nb1, nw2, nb2, hout);
}
